// GATEncoder_64974265254500
// MI455X (gfx1250) — matrix operand order repaired and checked by reading, not compiled, not run on hardware
//
#include <hip/hip_runtime.h>
#include <hip/hip_bf16.h>
#include <stdint.h>

// ---------------------------------------------------------------------------
// Types for CDNA5 WMMA (wave32): bf16 16x16x32 -> f32 accumulate
// ---------------------------------------------------------------------------
typedef __bf16 bf16_t;
typedef bf16_t bf16x8  __attribute__((ext_vector_type(8)));
typedef bf16_t bf16x16 __attribute__((ext_vector_type(16)));
typedef float  f32x8   __attribute__((ext_vector_type(8)));

#define DEVFN static __device__ __forceinline__

// Problem constants (match reference setup_inputs)
constexpr int NNODE = 4096;    // N
constexpr int GIN   = 3000;    // input features
constexpr int KP    = 3008;    // padded K (multiple of 32)
constexpr int D1    = 512;     // H*HID
constexpr int HIDC  = 128;     // per-head channels, layer 1
constexpr int NH    = 4;       // heads
constexpr int LATC  = 16;      // per-head channels, layers 2/3
constexpr int D2    = 64;      // H*LAT
constexpr int EEDG  = 131072;  // edges (without self loops)
constexpr int ETOT  = EEDG + NNODE;
constexpr unsigned NEG_INF_ENC = 0x007FFFFFu; // fenc(-inf)

DEVFN float lrelu(float x) { return x > 0.f ? x : 0.2f * x; }

// order-preserving float<->uint encoding for atomicMax on floats
DEVFN unsigned fenc(float f) {
  unsigned b = __float_as_uint(f);
  return ((int)b < 0) ? ~b : (b | 0x80000000u);
}
DEVFN float fdec(unsigned u) {
  return ((int)u < 0) ? __uint_as_float(u & 0x7fffffffu) : __uint_as_float(~u);
}

DEVFN bf16x16 cat16(bf16x8 lo, bf16x8 hi) {
  return __builtin_shufflevector(lo, hi, 0,1,2,3,4,5,6,7,8,9,10,11,12,13,14,15);
}
DEVFN f32x8 wmma_bf16(bf16x16 a, bf16x16 b, f32x8 c) {
  return __builtin_amdgcn_wmma_f32_16x16x32_bf16(false, a, false, b, (short)0, c, false, false);
}

// ---------------------------------------------------------------------------
// Small utility kernels
// ---------------------------------------------------------------------------
__global__ void fill_f32_kernel(float* p, int n, float v) {
  int i = blockIdx.x * blockDim.x + threadIdx.x;
  if (i < n) p[i] = v;
}
__global__ void fill_u32_kernel(unsigned* p, int n, unsigned v) {
  int i = blockIdx.x * blockDim.x + threadIdx.x;
  if (i < n) p[i] = v;
}
__global__ void bias_rows_kernel(float* out, const float* __restrict__ b, int rows, int D) {
  int i = blockIdx.x * blockDim.x + threadIdx.x;
  if (i < rows * D) out[i] = b[i % D];
}
__global__ void convert_bf16_kernel(const float* __restrict__ x, bf16_t* __restrict__ y, long long n) {
  long long i = (long long)blockIdx.x * blockDim.x + threadIdx.x;
  if (i < n) y[i] = (bf16_t)x[i];
}
// pad along columns: y[rows][colsP], zeros for c >= cols (A-side operands)
__global__ void convert_pad_cols_kernel(const float* __restrict__ x, bf16_t* __restrict__ y,
                                        int rows, int cols, int colsP) {
  long long i = (long long)blockIdx.x * blockDim.x + threadIdx.x;
  if (i >= (long long)rows * colsP) return;
  int r = (int)(i / colsP), c = (int)(i % colsP);
  y[i] = (c < cols) ? (bf16_t)x[(size_t)r * cols + c] : (bf16_t)0.f;
}
// fp32 X[R][Cc] -> bf16 Y[Cc][RP] transposed, rows r >= R zero padded (B-side operands)
__global__ void transpose_pad_bf16_kernel(const float* __restrict__ x, bf16_t* __restrict__ y,
                                          int R, int Cc, int RP) {
  long long i = (long long)blockIdx.x * blockDim.x + threadIdx.x;
  if (i >= (long long)Cc * RP) return;
  int c = (int)(i / RP), rp = (int)(i % RP);
  y[i] = (rp < R) ? (bf16_t)x[(size_t)rp * Cc + c] : (bf16_t)0.f;
}

// ---------------------------------------------------------------------------
// Tiled bf16 WMMA GEMM: C[M][N] (f32) = A[M][K] x BT[N][K]^T
// M,N % 64 == 0, K % 32 == 0. Block = 256 threads (8 waves), 64x64 C tile,
// wave -> 16x32 (2 WMMA tiles). Double-buffered LDS: global loads for tile
// i+1 are issued before the barrier so they overlap the WMMA work on tile i.
// ---------------------------------------------------------------------------
__global__ void gemm_bf16_kernel(const bf16_t* __restrict__ A, const bf16_t* __restrict__ BT,
                                 float* __restrict__ C, int M, int N, int K) {
  __shared__ __align__(16) bf16_t sA[2][64 * 32];   // [buf][row][k]
  __shared__ __align__(16) bf16_t sB[2][64 * 32];   // [buf][n][k]
  const int tid  = threadIdx.x;
  const int lane = tid & 31;
  const int wave = tid >> 5;
  const int hl   = lane >> 4;      // lane half
  const int r    = lane & 15;
  const int waveM = wave & 3, waveN = wave >> 2;
  const int m0 = blockIdx.y * 64, n0 = blockIdx.x * 64;

  f32x8 acc0 = {0,0,0,0,0,0,0,0};
  f32x8 acc1 = {0,0,0,0,0,0,0,0};

  const int idx = tid * 8;               // 2048 elems / 256 threads, 16B each
  const int row = idx >> 5, col = idx & 31;
  const bf16_t* gA = A  + (size_t)(m0 + row) * K + col;
  const bf16_t* gB = BT + (size_t)(n0 + row) * K + col;

  uint4 ra = *(const uint4*)gA;          // tile 0 in flight
  uint4 rb = *(const uint4*)gB;
  const int nk = K >> 5;

  for (int it = 0; it < nk; ++it) {
    const int cur = it & 1;
    *(uint4*)&sA[cur][idx] = ra;
    *(uint4*)&sB[cur][idx] = rb;
    if (it + 1 < nk) {                   // issue next tile's loads (overlaps compute)
      ra = *(const uint4*)(gA + (size_t)(it + 1) * 32);
      rb = *(const uint4*)(gB + (size_t)(it + 1) * 32);
    }
    __syncthreads();

    const bf16_t* a_ = sA[cur];
    const bf16_t* b_ = sB[cur];
    // A fragment: lane half 0 -> K 0..7 | 16..23 ; half 1 -> K 8..15 | 24..31
    const int rowW = (waveM * 16 + r) * 32;
    bf16x16 afr = cat16(*(const bf16x8*)&a_[rowW + hl * 8],
                        *(const bf16x8*)&a_[rowW + 16 + hl * 8]);
    // B fragments: lane half 0 -> K 0..7 | 16..23 ; half 1 -> K 8..15 | 24..31
    const int c0 = (waveN * 32 + r) * 32 + hl * 8;
    bf16x16 bf0 = cat16(*(const bf16x8*)&b_[c0], *(const bf16x8*)&b_[c0 + 16]);
    const int c1 = (waveN * 32 + 16 + r) * 32 + hl * 8;
    bf16x16 bf1 = cat16(*(const bf16x8*)&b_[c1], *(const bf16x8*)&b_[c1 + 16]);

    acc0 = wmma_bf16(afr, bf0, acc0);
    acc1 = wmma_bf16(afr, bf1, acc1);
    __syncthreads();
  }
  // C layout: VGPR v -> M = v + 8*half, N = lane%16
#pragma unroll
  for (int v = 0; v < 8; ++v) {
    int mr = m0 + waveM * 16 + v + hl * 8;
    C[(size_t)mr * N + n0 + waveN * 32 + r]      = acc0[v];
    C[(size_t)mr * N + n0 + waveN * 32 + 16 + r] = acc1[v];
  }
}

// ---------------------------------------------------------------------------
// e arrays are HEAD-MAJOR: e[h][n] (contiguous in n) for vectorized loads.
// e_src[h,n] = sum_c h[n,h,c]*a_src[h,c] ; e_dst likewise
// ---------------------------------------------------------------------------
__global__ void compute_e_kernel(const float* __restrict__ Hm, const float* __restrict__ aS,
                                 const float* __restrict__ aD, float* __restrict__ eS,
                                 float* __restrict__ eD, int C) {
  int t = blockIdx.x * blockDim.x + threadIdx.x;
  if (t >= NNODE * NH) return;
  int h = t / NNODE, n = t % NNODE;
  const float* row = Hm + (size_t)n * NH * C + h * C;
  float s = 0.f, d = 0.f;
  for (int c = 0; c < C; ++c) { s += row[c] * aS[h * C + c]; d += row[c] * aD[h * C + c]; }
  eS[t] = s; eD[t] = d;
}

// per-head max of e_src over all nodes (dense softmax closed-form row max,
// valid because leaky_relu is monotone)
__global__ void head_max_kernel(const float* __restrict__ eS, float* __restrict__ mxh) {
  __shared__ float red[256];
  int h = blockIdx.x;
  float m = -3.4e38f;
  for (int j = threadIdx.x; j < NNODE; j += 256) m = fmaxf(m, eS[h * NNODE + j]);
  red[threadIdx.x] = m;
  __syncthreads();
  for (int s = 128; s > 0; s >>= 1) {
    if (threadIdx.x < s) red[threadIdx.x] = fmaxf(red[threadIdx.x], red[threadIdx.x + s]);
    __syncthreads();
  }
  if (threadIdx.x == 0) mxh[h] = red[0];
}

// dense softmax denominators: denom[h,i] = sum_j exp(lr(eD_i + eS_j) - m_ih)
__global__ void dense_denom_kernel(const float* __restrict__ eS, const float* __restrict__ eD,
                                   const float* __restrict__ mxh, float* __restrict__ denom) {
  int t = blockIdx.x * blockDim.x + threadIdx.x;
  if (t >= NNODE * NH) return;
  int h = t / NNODE;
  float ed = eD[t];
  float m = lrelu(ed + mxh[h]);
  const float* pe = eS + (size_t)h * NNODE;
  float s = 0.f;
  for (int j = 0; j < NNODE; ++j) s += expf(lrelu(ed + pe[j]) - m);
  denom[t] = s;
}

// ---------------------------------------------------------------------------
// Fused dense attention: out[i, cols] = (P_i . H) / denom, P computed on the
// fly directly into WMMA A-fragments (flash style; alpha never materialized).
// HT = bf16 h transposed [D][NNODE]. NT = 16-col tiles per wave.
// Launch: layer1 D=512 -> NT=4, 256 thr; layers2/3 D=64 -> NT=1, 128 thr.
// ---------------------------------------------------------------------------
template <int NT>
__global__ void dense_attn_kernel(const bf16_t* __restrict__ HT, const float* __restrict__ eS,
                                  const float* __restrict__ eD, const float* __restrict__ mxh,
                                  const float* __restrict__ denom, float* __restrict__ out,
                                  int Dout, int Cph, const float* __restrict__ bias, int meanMode) {
  const int tid  = threadIdx.x;
  const int lane = tid & 31;
  const int wave = tid >> 5;
  const int hl   = lane >> 4;
  const int r    = lane & 15;
  const int colBase = wave * (NT * 16);
  const int head = colBase / Cph;
  const int I0 = blockIdx.x * 16;
  const int i = I0 + r;
  const float ed = eD[(size_t)head * NNODE + i];
  const float m  = lrelu(ed + mxh[head]);
  const float* pe = eS + (size_t)head * NNODE;

  f32x8 acc[NT];
#pragma unroll
  for (int t = 0; t < NT; ++t) { f32x8 z = {0,0,0,0,0,0,0,0}; acc[t] = z; }

  for (int kb = 0; kb < NNODE; kb += 32) {
    // P fragment in A-layout: row = r; lane-half K chunks 0..7|16..23 / 8..15|24..31
    const int j0 = kb + hl * 8;
    const int j1 = kb + 16 + hl * 8;
    float wv[16];
    *(float4*)&wv[0]  = *(const float4*)&pe[j0];
    *(float4*)&wv[4]  = *(const float4*)&pe[j0 + 4];
    *(float4*)&wv[8]  = *(const float4*)&pe[j1];
    *(float4*)&wv[12] = *(const float4*)&pe[j1 + 4];
    bf16x16 afr;
#pragma unroll
    for (int e = 0; e < 16; ++e)
      afr[e] = (bf16_t)expf(lrelu(ed + wv[e]) - m);

#pragma unroll
    for (int t = 0; t < NT; ++t) {
      const int col = colBase + t * 16 + r;
      const bf16_t* p = &HT[(size_t)col * NNODE + kb + hl * 8];
      bf16x16 bfr = cat16(*(const bf16x8*)p, *(const bf16x8*)(p + 16));
      acc[t] = wmma_bf16(afr, bfr, acc[t]);
    }
  }
#pragma unroll
  for (int t = 0; t < NT; ++t) {
#pragma unroll
    for (int v = 0; v < 8; ++v) {
      int i2 = I0 + v + hl * 8;
      int col = colBase + t * 16 + r;
      float val = acc[t][v] / denom[(size_t)head * NNODE + i2];
      if (meanMode)
        atomicAdd(&out[(size_t)i2 * Dout + (col - head * Cph)], val * (1.0f / NH));
      else
        out[(size_t)i2 * Dout + col] = val + bias[col];
    }
  }
}

// ---------------------------------------------------------------------------
// Sparse branch (edge list + self loops), segment softmax by dst via atomics
// ---------------------------------------------------------------------------
DEVFN void edge_sd(int idx, const int* __restrict__ src, const int* __restrict__ dst,
                   int& s, int& d) {
  if (idx < EEDG) { s = src[idx]; d = dst[idx]; } else { s = idx - EEDG; d = s; }
}

__global__ void sparse_max_kernel(const int* __restrict__ src, const int* __restrict__ dst,
                                  const float* __restrict__ eS, const float* __restrict__ eD,
                                  unsigned* __restrict__ mEnc) {
  int t = blockIdx.x * blockDim.x + threadIdx.x;
  if (t >= ETOT * NH) return;
  int idx = t >> 2, h = t & 3;
  int s, d; edge_sd(idx, src, dst, s, d);
  float e = lrelu(eS[h * NNODE + s] + eD[h * NNODE + d]);
  atomicMax(&mEnc[h * NNODE + d], fenc(e));
}

__global__ void sparse_denom_kernel(const int* __restrict__ src, const int* __restrict__ dst,
                                    const float* __restrict__ eS, const float* __restrict__ eD,
                                    const unsigned* __restrict__ mEnc, float* __restrict__ denomS) {
  int t = blockIdx.x * blockDim.x + threadIdx.x;
  if (t >= ETOT * NH) return;
  int idx = t >> 2, h = t & 3;
  int s, d; edge_sd(idx, src, dst, s, d);
  float e = lrelu(eS[h * NNODE + s] + eD[h * NNODE + d]);
  atomicAdd(&denomS[h * NNODE + d], expf(e - fdec(mEnc[h * NNODE + d])));
}

__global__ void sparse_agg_kernel(const int* __restrict__ src, const int* __restrict__ dst,
                                  const float* __restrict__ eS, const float* __restrict__ eD,
                                  const unsigned* __restrict__ mEnc, const float* __restrict__ denomS,
                                  const float* __restrict__ Hm, float* __restrict__ out,
                                  int C, int Dout, int meanMode) {
  int t = blockIdx.x * blockDim.x + threadIdx.x;
  if (t >= ETOT * NH) return;
  int idx = t >> 2, h = t & 3;
  int s, d; edge_sd(idx, src, dst, s, d);
  float e = lrelu(eS[h * NNODE + s] + eD[h * NNODE + d]);
  float alpha = expf(e - fdec(mEnc[h * NNODE + d])) / (denomS[h * NNODE + d] + 1e-16f);
  float sc = meanMode ? alpha * (1.0f / NH) : alpha;
  const float* hr = Hm + (size_t)s * NH * C + h * C;
  float* orow = out + (size_t)d * Dout + (meanMode ? 0 : h * C);
  for (int c = 0; c < C; ++c) atomicAdd(&orow[c], sc * hr[c]);
}

// ---------------------------------------------------------------------------
// Host orchestration
// ---------------------------------------------------------------------------
static inline dim3 g256(long long n) { return dim3((unsigned)((n + 255) / 256)); }

extern "C" void kernel_launch(void* const* d_in, const int* in_sizes, int n_in,
                              void* d_out, int out_size, void* d_ws, size_t ws_size,
                              hipStream_t stream) {
  (void)in_sizes; (void)n_in; (void)out_size; (void)ws_size;
  const float* x    = (const float*)d_in[0];
  const int*   ei   = (const int*)d_in[1];
  const int*   srcI = ei;
  const int*   dstI = ei + EEDG;
  const float* W1   = (const float*)d_in[3];
  const float* a1s  = (const float*)d_in[4];
  const float* a1d  = (const float*)d_in[5];
  const float* b1   = (const float*)d_in[6];
  const float* Wm   = (const float*)d_in[7];
  const float* ams  = (const float*)d_in[8];
  const float* amd2 = (const float*)d_in[9];
  const float* bm   = (const float*)d_in[10];
  const float* Ws   = (const float*)d_in[11];
  const float* ass  = (const float*)d_in[12];
  const float* asd  = (const float*)d_in[13];
  const float* bs   = (const float*)d_in[14];
  float* out = (float*)d_out;

  // bump allocator over workspace
  char* wsb = (char*)d_ws;
  size_t off = 0;
  auto alloc = [&](size_t bytes) -> void* {
    void* p = wsb + off;
    off = (off + bytes + 255) & ~(size_t)255;
    return p;
  };
  bf16_t*   xb    = (bf16_t*)alloc((size_t)NNODE * KP * 2);
  bf16_t*   W1bT  = (bf16_t*)alloc((size_t)D1 * KP * 2);     // transposed [N][K]
  bf16_t*   WmbT  = (bf16_t*)alloc((size_t)D2 * D1 * 2);     // transposed [N][K]
  bf16_t*   WsbT  = (bf16_t*)alloc((size_t)D2 * D1 * 2);     // transposed [N][K]
  float*    H1    = (float*)alloc((size_t)NNODE * D1 * 4);
  bf16_t*   H1T   = (bf16_t*)alloc((size_t)D1 * NNODE * 2);
  float*    e1s   = (float*)alloc((size_t)NH * NNODE * 4);   // head-major [H][N]
  float*    e1d   = (float*)alloc((size_t)NH * NNODE * 4);
  float*    mxh   = (float*)alloc(256);
  float*    denD  = (float*)alloc((size_t)NH * NNODE * 4);
  unsigned* mEnc  = (unsigned*)alloc((size_t)NH * NNODE * 4);
  float*    denS  = (float*)alloc((size_t)NH * NNODE * 4);
  float*    x1    = (float*)alloc((size_t)NNODE * D1 * 4);
  float*    x2    = (float*)alloc((size_t)NNODE * D1 * 4);
  bf16_t*   x1b   = (bf16_t*)alloc((size_t)NNODE * D1 * 2);
  bf16_t*   x2b   = (bf16_t*)alloc((size_t)NNODE * D1 * 2);
  float*    Hs    = (float*)alloc((size_t)NNODE * D2 * 4);
  bf16_t*   HsT   = (bf16_t*)alloc((size_t)D2 * NNODE * 2);
  float*    e2s   = (float*)alloc((size_t)NH * NNODE * 4);
  float*    e2d   = (float*)alloc((size_t)NH * NNODE * 4);

  // ---- convert inputs to bf16; B-side operands transposed to [N][K]
  convert_pad_cols_kernel<<<g256((long long)NNODE * KP), 256, 0, stream>>>(x, xb, NNODE, GIN, KP);
  transpose_pad_bf16_kernel<<<g256((long long)D1 * KP), 256, 0, stream>>>(W1, W1bT, GIN, D1, KP);
  transpose_pad_bf16_kernel<<<g256((long long)D2 * D1), 256, 0, stream>>>(Wm, WmbT, D1, D2, D1);
  transpose_pad_bf16_kernel<<<g256((long long)D2 * D1), 256, 0, stream>>>(Ws, WsbT, D1, D2, D1);

  // ---- h = x @ W1 (shared by both branches)
  gemm_bf16_kernel<<<dim3(D1 / 64, NNODE / 64), 256, 0, stream>>>(xb, W1bT, H1, NNODE, D1, KP);
  compute_e_kernel<<<g256(NNODE * NH), 256, 0, stream>>>(H1, a1s, a1d, e1s, e1d, HIDC);

  // ---- sparse layer 1 -> x1 (concat + b1)
  fill_u32_kernel<<<g256(NNODE * NH), 256, 0, stream>>>(mEnc, NNODE * NH, NEG_INF_ENC);
  fill_f32_kernel<<<g256(NNODE * NH), 256, 0, stream>>>(denS, NNODE * NH, 0.f);
  bias_rows_kernel<<<g256((long long)NNODE * D1), 256, 0, stream>>>(x1, b1, NNODE, D1);
  sparse_max_kernel<<<g256((long long)ETOT * NH), 256, 0, stream>>>(srcI, dstI, e1s, e1d, mEnc);
  sparse_denom_kernel<<<g256((long long)ETOT * NH), 256, 0, stream>>>(srcI, dstI, e1s, e1d, mEnc, denS);
  sparse_agg_kernel<<<g256((long long)ETOT * NH), 256, 0, stream>>>(srcI, dstI, e1s, e1d, mEnc, denS,
                                                                    H1, x1, HIDC, D1, 0);

  // ---- dense layer 1 -> x2 (fused softmax * h via WMMA)
  transpose_pad_bf16_kernel<<<g256((long long)NNODE * D1), 256, 0, stream>>>(H1, H1T, NNODE, D1, NNODE);
  head_max_kernel<<<NH, 256, 0, stream>>>(e1s, mxh);
  dense_denom_kernel<<<g256(NNODE * NH), 256, 0, stream>>>(e1s, e1d, mxh, denD);
  dense_attn_kernel<4><<<NNODE / 16, 256, 0, stream>>>(H1T, e1s, e1d, mxh, denD, x2, D1, HIDC, b1, 0);

  convert_bf16_kernel<<<g256((long long)NNODE * D1), 256, 0, stream>>>(x1, x1b, (long long)NNODE * D1);
  convert_bf16_kernel<<<g256((long long)NNODE * D1), 256, 0, stream>>>(x2, x2b, (long long)NNODE * D1);

  // ---- small sparse layer: X -> out (mean over heads + bias)
  auto small_sparse = [&](const bf16_t* Xb, const bf16_t* WbT, const float* aS, const float* aD,
                          const float* bvec, float* dstOut) {
    gemm_bf16_kernel<<<dim3(D2 / 64, NNODE / 64), 256, 0, stream>>>(Xb, WbT, Hs, NNODE, D2, D1);
    compute_e_kernel<<<g256(NNODE * NH), 256, 0, stream>>>(Hs, aS, aD, e2s, e2d, LATC);
    fill_u32_kernel<<<g256(NNODE * NH), 256, 0, stream>>>(mEnc, NNODE * NH, NEG_INF_ENC);
    fill_f32_kernel<<<g256(NNODE * NH), 256, 0, stream>>>(denS, NNODE * NH, 0.f);
    bias_rows_kernel<<<g256(NNODE * LATC), 256, 0, stream>>>(dstOut, bvec, NNODE, LATC);
    sparse_max_kernel<<<g256((long long)ETOT * NH), 256, 0, stream>>>(srcI, dstI, e2s, e2d, mEnc);
    sparse_denom_kernel<<<g256((long long)ETOT * NH), 256, 0, stream>>>(srcI, dstI, e2s, e2d, mEnc, denS);
    sparse_agg_kernel<<<g256((long long)ETOT * NH), 256, 0, stream>>>(srcI, dstI, e2s, e2d, mEnc, denS,
                                                                      Hs, dstOut, LATC, LATC, 1);
  };
  // ---- small dense layer: X -> out (mean over heads + bias)
  auto small_dense = [&](const bf16_t* Xb, const bf16_t* WbT, const float* aS, const float* aD,
                         const float* bvec, float* dstOut) {
    gemm_bf16_kernel<<<dim3(D2 / 64, NNODE / 64), 256, 0, stream>>>(Xb, WbT, Hs, NNODE, D2, D1);
    compute_e_kernel<<<g256(NNODE * NH), 256, 0, stream>>>(Hs, aS, aD, e2s, e2d, LATC);
    transpose_pad_bf16_kernel<<<g256((long long)NNODE * D2), 256, 0, stream>>>(Hs, HsT, NNODE, D2, NNODE);
    head_max_kernel<<<NH, 256, 0, stream>>>(e2s, mxh);
    dense_denom_kernel<<<g256(NNODE * NH), 256, 0, stream>>>(e2s, e2d, mxh, denD);
    bias_rows_kernel<<<g256(NNODE * LATC), 256, 0, stream>>>(dstOut, bvec, NNODE, LATC);
    dense_attn_kernel<1><<<NNODE / 16, 128, 0, stream>>>(HsT, e2s, e2d, mxh, denD, dstOut, LATC, LATC,
                                                         bvec, 1);
  };

  const int OSEG = NNODE * LATC;  // 65536
  small_sparse(x1b, WmbT, ams, amd2, bm, out + 0 * OSEG);   // z_mean1
  small_sparse(x1b, WsbT, ass, asd, bs, out + 1 * OSEG);    // z_log_std1
  small_dense (x2b, WmbT, ams, amd2, bm, out + 2 * OSEG);   // z_mean2
  small_dense (x2b, WsbT, ass, asd, bs, out + 3 * OSEG);    // z_log_std2
}
